// GlobalAttnBlock_29850022707537
// MI455X (gfx1250) — matrix operand order repaired and checked by reading, not compiled, not run on hardware
//
#include <hip/hip_runtime.h>
#include <hip/hip_bf16.h>

typedef __attribute__((ext_vector_type(16))) _Float16 v16h;
typedef __attribute__((ext_vector_type(8)))  _Float16 v8h;
typedef __attribute__((ext_vector_type(8)))  float    v8f;
typedef __attribute__((ext_vector_type(4)))  unsigned int v4u;
typedef __attribute__((ext_vector_type(8)))  int      v8i_;
typedef __attribute__((ext_vector_type(4)))  int      v4i_;

// ---------------------------------------------------------------------------
// WMMA helpers (CDNA5, wave32). D = A(16x32 f16) * B(32x16 f16) + C(16x16 f32)
// ---------------------------------------------------------------------------
__device__ __forceinline__ v8f wmma_f16(v16h a, v16h b, v8f c) {
  return __builtin_amdgcn_wmma_f32_16x16x32_f16(
      /*neg_a=*/false, a, /*neg_b=*/false, b,
      /*c_mod=*/(short)0, c, /*reuse_a=*/false, /*reuse_b=*/false);
}

// A operand: 16x32 tile from row-major [rows, ld] f16 storage.
// Lane layout (ISA 7.12.2): lane&15 = M row; halves 0..7 -> K = kb..kb+7,
// halves 8..15 -> K = kb+16..kb+23, kb = (lane>>4)*8.
__device__ __forceinline__ v16h load_a16x32(const _Float16* base, int ld,
                                            int row0, int k0) {
  int lane = threadIdx.x & 31;
  const _Float16* p =
      base + (size_t)(row0 + (lane & 15)) * ld + k0 + ((lane >> 4) << 3);
  v8h lo = *(const v8h*)p;
  v8h hi = *(const v8h*)(p + 16);
  v16h a;
#pragma unroll
  for (int i = 0; i < 8; ++i) { a[i] = lo[i]; a[i + 8] = hi[i]; }
  return a;
}

// B operand: 32x16 tile (KxN) read from *transposed* storage Bt[N][K] row-major.
// Lane layout: lane&15 = N col; halves 0..7 -> K = kb..kb+7,
// halves 8..15 -> K = kb+16..kb+23, kb = k0 + (lane>>4)*8.
__device__ __forceinline__ v16h load_bt32x16(const _Float16* base, int ld,
                                             int n0, int k0) {
  int lane = threadIdx.x & 31;
  const _Float16* p = base + (size_t)(n0 + (lane & 15)) * ld + k0 +
                      ((lane >> 4) << 3);
  v8h lo = *(const v8h*)p;
  v8h hi = *(const v8h*)(p + 16);
  v16h r;
#pragma unroll
  for (int i = 0; i < 8; ++i) { r[i] = lo[i]; r[i + 8] = hi[i]; }
  return r;
}

__device__ __forceinline__ float silu(float x) {
  return x / (1.0f + __expf(-x));
}

#define B_  4
#define N_  1024
#define HID_ 256
#define NH_ 8
#define DH_ 32
#define BH_ 64

// ---------------------------------------------------------------------------
// 0) Convert & transpose weights to f16:  WqkvT[768][256], WprojT[256][256],
//    Wb2T[64][64], Wb3T[16][64] (rows 8..15 zero-padded).
// ---------------------------------------------------------------------------
__global__ __launch_bounds__(256) void convw_kernel(
    const float* __restrict__ Wqkv, const float* __restrict__ Wproj,
    const float* __restrict__ Wb2, const float* __restrict__ Wb3,
    _Float16* __restrict__ WqkvT, _Float16* __restrict__ WprojT,
    _Float16* __restrict__ Wb2T, _Float16* __restrict__ Wb3T) {
  int tid = blockIdx.x * 256 + threadIdx.x;
  if (tid < 196608) {                    // 768*256
    int o = tid >> 8, i = tid & 255;
    WqkvT[tid] = (_Float16)Wqkv[i * 768 + o];
  } else if (tid < 262144) {             // + 256*256
    int t = tid - 196608;
    int o = t >> 8, i = t & 255;
    WprojT[t] = (_Float16)Wproj[i * 256 + o];
  } else if (tid < 266240) {             // + 64*64
    int t = tid - 262144;
    int o = t >> 6, i = t & 63;
    Wb2T[t] = (_Float16)Wb2[i * 64 + o];
  } else if (tid < 267264) {             // + 16*64 (padded)
    int t = tid - 266240;
    int o = t >> 6, i = t & 63;
    Wb3T[t] = (o < 8) ? (_Float16)Wb3[i * 8 + o] : (_Float16)0.0f;
  }
}

// ---------------------------------------------------------------------------
// 1) LayerNorm: one wave per row of h[4096][256] -> xln f16
// ---------------------------------------------------------------------------
__global__ __launch_bounds__(256) void ln_kernel(
    const float* __restrict__ h, const float* __restrict__ gamma,
    const float* __restrict__ beta, _Float16* __restrict__ xln) {
  int wave = threadIdx.x >> 5, lane = threadIdx.x & 31;
  int row = blockIdx.x * 8 + wave;
  const float* x = h + (size_t)row * HID_;
  float v[8], s = 0.f, s2 = 0.f;
#pragma unroll
  for (int i = 0; i < 8; ++i) {
    v[i] = x[lane + i * 32];
    s += v[i];
    s2 += v[i] * v[i];
  }
#pragma unroll
  for (int off = 16; off > 0; off >>= 1) {
    s += __shfl_xor(s, off);
    s2 += __shfl_xor(s2, off);
  }
  float mu = s * (1.0f / HID_);
  float var = s2 * (1.0f / HID_) - mu * mu;
  float r = rsqrtf(var + 1e-5f);
#pragma unroll
  for (int i = 0; i < 8; ++i) {
    int c = lane + i * 32;
    xln[(size_t)row * HID_ + c] =
        (_Float16)((v[i] - mu) * r * gamma[c] + beta[c]);
  }
}

// ---------------------------------------------------------------------------
// 2) QKV GEMM: [4096,256]f16 x WqkvT -> scatter to q/k [B,NH,N,DH] and
//    transposed v [B,NH,DH,N].  One wave per 16x64 tile: A reused for 4 WMMAs
//    per K-step (4x arithmetic intensity vs 16x16 tiles).
// ---------------------------------------------------------------------------
__global__ __launch_bounds__(32) void qkv_kernel(
    const _Float16* __restrict__ xln, const _Float16* __restrict__ WqkvT,
    const float* __restrict__ bqkv, _Float16* __restrict__ qf,
    _Float16* __restrict__ kf, _Float16* __restrict__ vT) {
  int lane = threadIdx.x;
  int nt = blockIdx.x;   // 0..11 (768/64)
  int mt = blockIdx.y;   // 0..255 (4096/16)
  v8f acc[4] = {v8f{}, v8f{}, v8f{}, v8f{}};
#pragma unroll
  for (int ks = 0; ks < 8; ++ks) {
    v16h a = load_a16x32(xln, HID_, mt * 16, ks * 32);
#pragma unroll
    for (int t = 0; t < 4; ++t) {
      v16h b = load_bt32x16(WqkvT, HID_, nt * 64 + t * 16, ks * 32);
      acc[t] = wmma_f16(a, b, acc[t]);
    }
  }
#pragma unroll
  for (int t = 0; t < 4; ++t) {
    int c = nt * 64 + t * 16 + (lane & 15);
    float bv = bqkv[c];
    int sel = c >> 8;          // 0=q 1=k 2=v
    int rem = c & 255;
    int head = rem >> 5, d = rem & 31;
#pragma unroll
    for (int j = 0; j < 8; ++j) {
      int r = mt * 16 + j + ((lane >> 4) << 3);
      int b_ = r >> 10, n = r & 1023;
      _Float16 val = (_Float16)(acc[t][j] + bv);
      size_t bh = (size_t)(b_ * NH_ + head);
      if (sel == 0)      qf[(bh * N_ + n) * DH_ + d] = val;
      else if (sel == 1) kf[(bh * N_ + n) * DH_ + d] = val;
      else               vT[(bh * DH_ + d) * N_ + n] = val;
    }
  }
}

// ---------------------------------------------------------------------------
// 3) Relative-bias MLP: per (b, qt, kt) 16x16 pair tile (256 pairs).
//    L1 (K=1) in VALU, L2 [256,64]x[64,64] and L3 [256,64]x[64,16] via WMMA.
//    Wb2T/Wb3T held entirely in VGPRs across the Mtile loop.
// ---------------------------------------------------------------------------
__global__ __launch_bounds__(32) void bias_mlp_kernel(
    const float* __restrict__ coord, const float* __restrict__ Wb1,
    const float* __restrict__ bb1, const float* __restrict__ bb2,
    const float* __restrict__ bb3, const _Float16* __restrict__ Wb2T,
    const _Float16* __restrict__ Wb3T, _Float16* __restrict__ biasf16) {
  __shared__ _Float16 t1[256 * BH_];   // layer-1 activations (32 KB)
  __shared__ _Float16 t2[16 * BH_];    // per-Mtile layer-2 activations (2 KB)
  __shared__ float w1s[BH_], c1s[BH_];

  int lane = threadIdx.x;
  int kt = blockIdx.x, qt = blockIdx.y, b = blockIdx.z;

  w1s[lane] = Wb1[lane];   w1s[lane + 32] = Wb1[lane + 32];
  c1s[lane] = bb1[lane];   c1s[lane + 32] = bb1[lane + 32];

  // ---- layer 1: t1[p][j] = silu(dcoord_p * Wb1[j] + bb1[j]) --------------
#pragma unroll
  for (int pi = 0; pi < 8; ++pi) {
    int p = lane * 8 + pi;
    int qi = p >> 4, kj = p & 15;
    float d = coord[b * N_ + kt * 16 + kj] - coord[b * N_ + qt * 16 + qi];
#pragma unroll
    for (int j = 0; j < BH_; ++j)
      t1[p * BH_ + j] = (_Float16)silu(d * w1s[j] + c1s[j]);
  }

  float bb2v[4];
#pragma unroll
  for (int nt = 0; nt < 4; ++nt) bb2v[nt] = bb2[nt * 16 + (lane & 15)];
  float bb3v = ((lane & 15) < NH_) ? bb3[lane & 15] : 0.0f;

  // hoist all layer-2/3 weights into VGPRs (80 VGPRs, reused 16x)
  v16h w2[4][2];
#pragma unroll
  for (int nt = 0; nt < 4; ++nt) {
    w2[nt][0] = load_bt32x16(Wb2T, BH_, nt * 16, 0);
    w2[nt][1] = load_bt32x16(Wb2T, BH_, nt * 16, 32);
  }
  v16h w3_0 = load_bt32x16(Wb3T, BH_, 0, 0);
  v16h w3_1 = load_bt32x16(Wb3T, BH_, 0, 32);

  for (int mt = 0; mt < 16; ++mt) {
    // ---- layer 2: [16,64] x [64,64] -> silu -> t2 -----------------------
    v16h a0 = load_a16x32(t1 + mt * 16 * BH_, BH_, 0, 0);
    v16h a1 = load_a16x32(t1 + mt * 16 * BH_, BH_, 0, 32);
#pragma unroll
    for (int nt = 0; nt < 4; ++nt) {
      v8f acc = {};
      acc = wmma_f16(a0, w2[nt][0], acc);
      acc = wmma_f16(a1, w2[nt][1], acc);
      int n = nt * 16 + (lane & 15);
#pragma unroll
      for (int j = 0; j < 8; ++j) {
        int m = j + ((lane >> 4) << 3);
        t2[m * BH_ + n] = (_Float16)silu(acc[j] + bb2v[nt]);
      }
    }
    // ---- layer 3: [16,64] x [64,16pad] -> bias for 8 heads --------------
    v8f acc3 = {};
    v16h a20 = load_a16x32(t2, BH_, 0, 0);
    v16h a21 = load_a16x32(t2, BH_, 0, 32);
    acc3 = wmma_f16(a20, w3_0, acc3);
    acc3 = wmma_f16(a21, w3_1, acc3);
    int n = lane & 15;
    if (n < NH_) {
#pragma unroll
      for (int j = 0; j < 8; ++j) {
        int m = j + ((lane >> 4) << 3);
        int p = mt * 16 + m;
        int q = qt * 16 + (p >> 4);
        int k = kt * 16 + (p & 15);
        biasf16[((size_t)(b * NH_ + n) * N_ + q) * N_ + k] =
            (_Float16)(acc3[j] + bb3v);
      }
    }
  }
}

// ---------------------------------------------------------------------------
// 4) Attention: per (b, h, 16-row q tile).
//    The 16x1024 f16 bias slab is DMA'd into LDS by the Tensor Data Mover
//    (TENSOR_LOAD_TO_LDS, tracked by TENSORcnt), overlapping the score WMMAs'
//    operand loads; scores go to a 64KB LDS buffer, two-pass softmax, then
//    w@v WMMA with on-the-fly f32->f16 A build.
// ---------------------------------------------------------------------------
__global__ __launch_bounds__(32) void attn_kernel(
    const _Float16* __restrict__ qf, const _Float16* __restrict__ kf,
    const _Float16* __restrict__ vT, const _Float16* __restrict__ biasf16,
    const int* __restrict__ mask, _Float16* __restrict__ attnout) {
  __shared__ float sc[16 * N_];          // 64 KB score buffer
  __shared__ _Float16 biasLds[16 * N_];  // 32 KB TDM-staged bias slab
  int lane = threadIdx.x;
  int qt = blockIdx.x, h = blockIdx.y, b = blockIdx.z;
  size_t bh = (size_t)(b * NH_ + h);
  const _Float16* qbase = qf + bh * N_ * DH_;
  const _Float16* kbase = kf + bh * N_ * DH_;
  const _Float16* vbase = vT + bh * DH_ * N_;
  const float scale = 0.17677669529663687f;   // 1/sqrt(32)

  // ---- issue TDM: 16 rows x 1024 cols of f16 bias -> LDS -----------------
  {
    unsigned int ldsAddr = (unsigned int)(unsigned long long)(&biasLds[0]);
    unsigned long long ga =
        (unsigned long long)(const void*)(biasf16 + (bh * N_ + (size_t)qt * 16) * N_);
    v4u g0;
    g0[0] = 1u;                                   // count=1, user mode
    g0[1] = ldsAddr;                              // lds_addr
    g0[2] = (unsigned int)(ga & 0xffffffffu);     // global_addr[31:0]
    g0[3] = (unsigned int)((ga >> 32) & 0x01ffffffu) | (2u << 30);  // type=2
    v8i_ g1;
    g1[0] = (int)(1u << 16);                      // data_size=1 (2 bytes)
    g1[1] = (int)((N_ & 0xffffu) << 16);          // tensor_dim0 = 1024 (lo16)
    g1[2] = (int)((16u & 0xffffu) << 16);         // dim0 hi=0 | tensor_dim1=16
    g1[3] = (int)((unsigned)N_ << 16);            // dim1 hi=0 | tile_dim0=1024
    g1[4] = (int)16;                              // tile_dim1=16, tile_dim2=0
    g1[5] = (int)N_;                              // tensor_dim0_stride = 1024
    g1[6] = 0;
    g1[7] = 0;
    v4i_ g2 = {0, 0, 0, 0};
    v4i_ g3 = {0, 0, 0, 0};
#if defined(__clang_major__) && (__clang_major__ >= 23)
    v8i_ g4 = {0, 0, 0, 0, 0, 0, 0, 0};
    __builtin_amdgcn_tensor_load_to_lds(g0, g1, g2, g3, g4, 0);
#else
    __builtin_amdgcn_tensor_load_to_lds(g0, g1, g2, g3, 0);
#endif
  }

  v16h qa = load_a16x32(qbase, DH_, qt * 16, 0);

  // wait for the bias slab before first use
  __builtin_amdgcn_s_wait_tensorcnt((short)0);

  for (int kt = 0; kt < 64; ++kt) {
    if (kt < 63) {   // prefetch next k tile (global_prefetch_b8)
      __builtin_prefetch(kbase + (size_t)(kt + 1) * 16 * DH_, 0, 0);
    }
    v16h kb = load_bt32x16(kbase, DH_, kt * 16, 0);
    v8f s = {};
    s = wmma_f16(qa, kb, s);
    int kcol = kt * 16 + (lane & 15);
    int mk = mask[b * N_ + kcol];
#pragma unroll
    for (int j = 0; j < 8; ++j) {
      int m = j + ((lane >> 4) << 3);
      float bias = (float)biasLds[m * N_ + kcol];
      float val = s[j] * scale + bias;
      if (mk == 0) val = -__builtin_inff();
      sc[m * N_ + kcol] = val;
    }
  }

  // softmax: 2 lanes per row, 512 columns each
  int row = lane >> 1, half = lane & 1;
  float mx = -__builtin_inff();
  for (int i = 0; i < 512; ++i)
    mx = fmaxf(mx, sc[row * N_ + half * 512 + i]);
  mx = fmaxf(mx, __shfl_xor(mx, 1));
  float sum = 0.f;
  for (int i = 0; i < 512; ++i) {
    int idx = row * N_ + half * 512 + i;
    float e = __expf(sc[idx] - mx);
    sc[idx] = e;
    sum += e;
  }
  sum += __shfl_xor(sum, 1);
  float inv = 1.0f / sum;   // valid in both lanes of the pair

  // w@v: K = 1024 in 32 WMMA steps, two 16-wide N tiles (DH = 32)
  int m = lane & 15;
  int kb8 = (lane >> 4) << 3;
  float isv = __shfl(inv, (lane & 15) * 2);   // row-m scaling factor
  v8f acc0 = {}, acc1 = {};
  for (int ks = 0; ks < 32; ++ks) {
    if (ks < 31) {   // prefetch next v K-slab
      __builtin_prefetch(vbase + (size_t)(ks + 1) * 32, 0, 0);
    }
    v16h a;
#pragma unroll
    for (int i = 0; i < 8; ++i) {
      a[i]     = (_Float16)(sc[m * N_ + ks * 32 + kb8 + i] * isv);
      a[i + 8] = (_Float16)(sc[m * N_ + ks * 32 + kb8 + 16 + i] * isv);
    }
    v16h b0 = load_bt32x16(vbase, N_, 0, ks * 32);
    v16h b1 = load_bt32x16(vbase, N_, 16, ks * 32);
    acc0 = wmma_f16(a, b0, acc0);
    acc1 = wmma_f16(a, b1, acc1);
  }
#pragma unroll
  for (int j = 0; j < 8; ++j) {
    int mm = j + ((lane >> 4) << 3);
    int qrow = qt * 16 + mm;
    int d = lane & 15;
    size_t base = ((size_t)(b * N_ + qrow)) * HID_ + h * DH_;
    attnout[base + d] = (_Float16)acc0[j];
    attnout[base + 16 + d] = (_Float16)acc1[j];
  }
}

// ---------------------------------------------------------------------------
// 5) Output projection + residual: out = h + attn @ Wproj + bproj (f32 out)
//    16x64 tile per wave (A reuse x4).
// ---------------------------------------------------------------------------
__global__ __launch_bounds__(32) void proj_kernel(
    const _Float16* __restrict__ attnf, const _Float16* __restrict__ WprojT,
    const float* __restrict__ bproj, const float* __restrict__ h,
    float* __restrict__ out) {
  int lane = threadIdx.x;
  int nt = blockIdx.x;   // 0..3  (256/64)
  int mt = blockIdx.y;   // 0..255
  v8f acc[4] = {v8f{}, v8f{}, v8f{}, v8f{}};
#pragma unroll
  for (int ks = 0; ks < 8; ++ks) {
    v16h a = load_a16x32(attnf, HID_, mt * 16, ks * 32);
#pragma unroll
    for (int t = 0; t < 4; ++t) {
      v16h b = load_bt32x16(WprojT, HID_, nt * 64 + t * 16, ks * 32);
      acc[t] = wmma_f16(a, b, acc[t]);
    }
  }
#pragma unroll
  for (int t = 0; t < 4; ++t) {
    int c = nt * 64 + t * 16 + (lane & 15);
    float bv = bproj[c];
#pragma unroll
    for (int j = 0; j < 8; ++j) {
      int r = mt * 16 + j + ((lane >> 4) << 3);
      out[(size_t)r * HID_ + c] = acc[t][j] + bv + h[(size_t)r * HID_ + c];
    }
  }
}

// ---------------------------------------------------------------------------
extern "C" void kernel_launch(void* const* d_in, const int* in_sizes, int n_in,
                              void* d_out, int out_size, void* d_ws,
                              size_t ws_size, hipStream_t stream) {
  const float* h     = (const float*)d_in[0];
  const float* coord = (const float*)d_in[1];
  const int*   mask  = (const int*)d_in[2];
  const float* Wqkv  = (const float*)d_in[3];
  const float* bqkv  = (const float*)d_in[4];
  const float* Wproj = (const float*)d_in[5];
  const float* bproj = (const float*)d_in[6];
  const float* gamma = (const float*)d_in[7];
  const float* beta  = (const float*)d_in[8];
  const float* Wb1   = (const float*)d_in[9];
  const float* bb1   = (const float*)d_in[10];
  const float* Wb2   = (const float*)d_in[11];
  const float* bb2   = (const float*)d_in[12];
  const float* Wb3   = (const float*)d_in[13];
  const float* bb3   = (const float*)d_in[14];
  float* out = (float*)d_out;

  char* ws = (char*)d_ws;
  size_t off = 0;
  auto alloc = [&](size_t bytes) -> void* {
    void* p = ws + off;
    off = (off + bytes + 255) & ~(size_t)255;
    return p;
  };
  _Float16* xln    = (_Float16*)alloc((size_t)B_ * N_ * HID_ * 2);        // 2 MB
  _Float16* qf     = (_Float16*)alloc((size_t)B_ * NH_ * N_ * DH_ * 2);   // 2 MB
  _Float16* kfm    = (_Float16*)alloc((size_t)B_ * NH_ * N_ * DH_ * 2);   // 2 MB
  _Float16* vT     = (_Float16*)alloc((size_t)B_ * NH_ * DH_ * N_ * 2);   // 2 MB
  _Float16* attnf  = (_Float16*)alloc((size_t)B_ * N_ * HID_ * 2);        // 2 MB
  _Float16* WqkvT  = (_Float16*)alloc((size_t)768 * 256 * 2);
  _Float16* WprojT = (_Float16*)alloc((size_t)256 * 256 * 2);
  _Float16* Wb2T   = (_Float16*)alloc((size_t)64 * 64 * 2);
  _Float16* Wb3T   = (_Float16*)alloc((size_t)16 * 64 * 2);
  _Float16* biasW  = (_Float16*)alloc((size_t)B_ * NH_ * N_ * N_ * 2);    // 64 MB

  convw_kernel<<<1044, 256, 0, stream>>>(Wqkv, Wproj, Wb2, Wb3,
                                         WqkvT, WprojT, Wb2T, Wb3T);
  ln_kernel<<<512, 256, 0, stream>>>(h, gamma, beta, xln);
  qkv_kernel<<<dim3(12, 256), 32, 0, stream>>>(xln, WqkvT, bqkv, qf, kfm, vT);
  bias_mlp_kernel<<<dim3(64, 64, 4), 32, 0, stream>>>(coord, Wb1, bb1, bb2,
                                                      bb3, Wb2T, Wb3T, biasW);
  attn_kernel<<<dim3(64, 8, 4), 32, 0, stream>>>(qf, kfm, vT, biasW, mask,
                                                 attnf);
  proj_kernel<<<dim3(4, 256), 32, 0, stream>>>(attnf, WprojT, bproj, h, out);
}
